// BiLSTMLayer_31868657336665
// MI455X (gfx1250) — matrix operand order repaired and checked by reading, not compiled, not run on hardware
//
#include <hip/hip_runtime.h>
#include <hip/hip_bf16.h>
#include <stdint.h>

typedef __bf16 bf16raw;
typedef bf16raw v16bf __attribute__((ext_vector_type(16)));
typedef float   v8f   __attribute__((ext_vector_type(8)));
typedef int     v4i_vs __attribute__((vector_size(16)));    // b128 payload type
typedef unsigned int u32x4 __attribute__((vector_size(16)));
typedef int i32x8 __attribute__((vector_size(32)));
typedef int i32x4 __attribute__((vector_size(16)));

union Frag32B { uint4 q[2]; v16bf v; };

#define BATCH 32
#define TLEN  256
#define DIM   1024   // D == H
#define NGATE 4096   // 4*H
#define NWG_DIR 16

#if defined(__has_builtin)
#if __has_builtin(__builtin_amdgcn_global_load_async_to_lds_b128)
#define HAVE_ASYNC_LDS 1
#endif
#if __has_builtin(__builtin_amdgcn_tensor_load_to_lds)
#define HAVE_TDM 1
#endif
#endif

__device__ __forceinline__ v8f wmma_bf16(v16bf a, v16bf b, v8f c) {
  // D = A(16x32 bf16) * B(32x16 bf16) + C(16x16 f32)
  return __builtin_amdgcn_wmma_f32_16x16x32_bf16(
      /*neg_a=*/false, a, /*neg_b=*/false, b,
      /*c_mod=*/(short)0, c, /*reuse_a=*/false, /*reuse_b=*/false);
}

// A fragment: 16x32 (MxK) bf16, row-major source with K-contiguous rows.
__device__ __forceinline__ v16bf load_a_frag(const __hip_bfloat16* base,
                                             int m, int ko, int lane) {
  const int off = (lane & 16) ? 16 : 0;
  const __hip_bfloat16* p = base + (size_t)m * DIM + ko + off;
  Frag32B f;
  f.q[0] = *(const uint4*)(p);
  f.q[1] = *(const uint4*)(p + 8);
  return f.v;
}

// B fragment: 32x16 (KxN) bf16 from a transposed [n][k] matrix.
__device__ __forceinline__ v16bf load_b_frag(const __hip_bfloat16* bT,
                                             int n, int ko, int lane) {
  const __hip_bfloat16* p = bT + (size_t)n * DIM + ko + ((lane & 16) ? 16 : 0);
  Frag32B f;
  f.q[0] = *(const uint4*)(p);
  f.q[1] = *(const uint4*)(p + 8);
  return f.v;
}

__device__ __forceinline__ float sigmoidf_(float v) {
  return 1.0f / (1.0f + __expf(-v));
}

// ---------------------------------------------------------------------------
// Kernel 1: pack f32 inputs to bf16 (Xb row-major; WxT/WhT transposed [n][k]).
// gate order g: 0=f 1=i 2=c 3=o
// ---------------------------------------------------------------------------
__global__ __launch_bounds__(256) void pack_kernel(
    const float* __restrict__ x,
    const float* __restrict__ Wf, const float* __restrict__ Wi,
    const float* __restrict__ Wc, const float* __restrict__ Wo,
    __hip_bfloat16* __restrict__ Xb,
    __hip_bfloat16* __restrict__ WxT,
    __hip_bfloat16* __restrict__ WhT) {
  const size_t idx = (size_t)blockIdx.x * 256 + threadIdx.x;
  if (idx < (size_t)BATCH * TLEN * DIM) {
    Xb[idx] = __float2bfloat16(x[idx]);
  }
  if (idx < (size_t)NGATE * DIM) {
    const int n = (int)(idx >> 10);
    const int k = (int)(idx & 1023);
    const int g = n >> 10;
    const int col = n & 1023;
    const float* W = (g == 0) ? Wf : (g == 1) ? Wi : (g == 2) ? Wc : Wo;
    WxT[idx] = __float2bfloat16(W[(size_t)(DIM + k) * DIM + col]);
    WhT[idx] = __float2bfloat16(W[(size_t)k * DIM + col]);
  }
}

// ---------------------------------------------------------------------------
// Kernel 2: pre[bt][n] = (X @ Wx)[bt][n] + bias[n]   (f32, n gate-major)
// One wave: 16(M) x 64(N) tile, K=1024, two-stage ping-pong pipeline
// (fragments reloaded in place -> no cross-set register copies).
// ---------------------------------------------------------------------------
__global__ __launch_bounds__(256) void gemm_pre_kernel(
    const __hip_bfloat16* __restrict__ Xb,
    const __hip_bfloat16* __restrict__ WxT,
    const float* __restrict__ b0, const float* __restrict__ b1,
    const float* __restrict__ b2, const float* __restrict__ b3,
    float* __restrict__ pre) {
  const int lane = threadIdx.x & 31;
  const int wid  = (int)((blockIdx.x * 256 + threadIdx.x) >> 5);
  const int nt   = wid & 63;    // 64-wide N tile
  const int mt   = wid >> 6;    // 16-high M tile, 0..511
  const int mbase = mt << 4;
  const int nbase = nt << 6;
  const int am = mbase + (lane & 15);
  const int nc = lane & 15;

  v8f acc[4] = {};
  v16bf aA = load_a_frag(Xb, am, 0, lane);
  v16bf bA[4];
#pragma unroll
  for (int j = 0; j < 4; ++j) bA[j] = load_b_frag(WxT, nbase + j * 16 + nc, 0, lane);

  for (int ko = 0; ko < DIM; ko += 64) {
    // stage 2 fragments (ko+32 always < DIM since DIM % 64 == 0)
    v16bf aB = load_a_frag(Xb, am, ko + 32, lane);
    v16bf bB[4];
#pragma unroll
    for (int j = 0; j < 4; ++j) bB[j] = load_b_frag(WxT, nbase + j * 16 + nc, ko + 32, lane);
#pragma unroll
    for (int j = 0; j < 4; ++j) acc[j] = wmma_bf16(aA, bA[j], acc[j]);
    if (ko + 64 < DIM) {   // reload stage-1 set in place for next iteration
      aA = load_a_frag(Xb, am, ko + 64, lane);
#pragma unroll
      for (int j = 0; j < 4; ++j) bA[j] = load_b_frag(WxT, nbase + j * 16 + nc, ko + 64, lane);
    }
#pragma unroll
    for (int j = 0; j < 4; ++j) acc[j] = wmma_bf16(aB, bB[j], acc[j]);
  }

  const int gate = nbase >> 10;  // uniform per wave (64 | 1024)
  const float* bias = (gate == 0) ? b0 : (gate == 1) ? b1 : (gate == 2) ? b2 : b3;
  const int mrow = mbase + ((lane >> 4) << 3);
#pragma unroll
  for (int j = 0; j < 4; ++j) {
    const int n = nbase + j * 16 + nc;
    const float bv = bias[n & 1023];
#pragma unroll
    for (int r = 0; r < 8; ++r) {
      pre[(size_t)(mrow + r) * NGATE + n] = acc[j][r] + bv;
    }
  }
}

// ---------------------------------------------------------------------------
// Kernel 3: persistent bidirectional recurrence.
// 32 WGs: dir = blockIdx.x>>4, 16 WGs/dir, each owns 64 h-cols x 4 gates x 32 b.
// Per step: TDM-stage h (64KB) to LDS, async-gather this step's 32KB pre slice
// to LDS (overlapped with the WMMA K-loop), combine gates per-lane.
// ---------------------------------------------------------------------------
__device__ __forceinline__ void grid_barrier(volatile unsigned* ctr, unsigned target) {
  __syncthreads();
  if (threadIdx.x == 0) {
    __threadfence();
    atomicAdd((unsigned*)ctr, 1u);
    while (*ctr < target) __builtin_amdgcn_s_sleep(1);
    __threadfence();
  }
  __syncthreads();
}

__global__ __launch_bounds__(256) void lstm_recurrent_kernel(
    const __hip_bfloat16* __restrict__ WhT,
    const float* __restrict__ pre,
    __hip_bfloat16* __restrict__ hstate,   // [2][32][1024] bf16
    float* __restrict__ cstate,            // [2][32][1024] f32
    unsigned* __restrict__ ctrs,
    float* __restrict__ out) {             // [B][T][2H] f32
  __shared__ __align__(16) __hip_bfloat16 shH[BATCH * DIM];   // 64 KB
  __shared__ __align__(16) float shPre[BATCH * 4 * 64];       // 32 KB

  const int lane = threadIdx.x & 31;
  const int w    = threadIdx.x >> 5;       // 8 waves
  const int dir  = blockIdx.x >> 4;
  const int wg   = blockIdx.x & 15;
  const int mt   = w >> 2;                 // 0..1 : batch 16-row tile
  const int colt = w & 3;                  // 0..3 : 16-col sub-tile
  const int col16 = (wg << 6) + (colt << 4);

  __hip_bfloat16* hs = hstate + (size_t)dir * BATCH * DIM;
  float*          cs = cstate + (size_t)dir * BATCH * DIM;
  volatile unsigned* ctr = ctrs + dir * 32;

  unsigned target = 0;

  for (int t = 0; t < TLEN; ++t) {
    const int ts = dir ? (TLEN - 1 - t) : t;

    // ---- 1) async-gather this step's pre-activation slice into LDS ----
    // chunk c (16B = 4 floats): m = c>>6, gate = (c>>4)&3, cols (c&15)*4..+3
#ifdef HAVE_ASYNC_LDS
#pragma unroll
    for (int k8 = 0; k8 < 8; ++k8) {
      const int c  = (int)threadIdx.x * 8 + k8;
      const int m  = c >> 6;
      const int g  = (c >> 4) & 3;
      const int cc = (c & 15) << 2;
      const float* gsrc = pre + ((size_t)m * TLEN + ts) * NGATE + (g << 10) + (wg << 6) + cc;
      float* ldst = shPre + ((m << 2) + g) * 64 + cc;
      __builtin_amdgcn_global_load_async_to_lds_b128((v4i_vs*)gsrc, (v4i_vs*)ldst, 0, 0);
    }
#else
#pragma unroll
    for (int k8 = 0; k8 < 8; ++k8) {
      const int c  = (int)threadIdx.x * 8 + k8;
      const int m  = c >> 6;
      const int g  = (c >> 4) & 3;
      const int cc = (c & 15) << 2;
      const float* gsrc = pre + ((size_t)m * TLEN + ts) * NGATE + (g << 10) + (wg << 6) + cc;
      *(uint4*)(shPre + ((m << 2) + g) * 64 + cc) = *(const uint4*)gsrc;
    }
#endif

    // ---- 2) stage h_state -> LDS (64 KB contiguous): Tensor Data Mover ----
#ifdef HAVE_TDM
    if (threadIdx.x < 32) {   // one wave issues the DMA
      const unsigned long long ga = (unsigned long long)(uintptr_t)hs;
      const unsigned lds_off = (unsigned)(uintptr_t)(&shH[0]);
      // D# group0: count=1 | lds_addr | global_addr | type=2
      u32x4 g0 = { 1u, lds_off,
                   (unsigned)(ga & 0xFFFFFFFFu),
                   (unsigned)((ga >> 32) & 0x01FFFFFFu) | (2u << 30) };
      // D# group1: data_size=2B; tensor_dim0=32768; tensor_dim1=1;
      //            tile_dim0=32768; tile_dim1=1; dim0_stride=32768
      i32x8 g1 = { (int)0x00010000,   // wg_mask=0, data_size=1 (2B)
                   (int)0x80000000,   // tensor_dim0 lo16 << 16
                   (int)0x00010000,   // tensor_dim0 hi16 | tensor_dim1 lo16<<16
                   (int)0x80000000,   // tensor_dim1 hi | tile_dim0<<16
                   (int)0x00000001,   // tile_dim1=1, tile_dim2=0
                   (int)0x00008000,   // tensor_dim0_stride lo32 = 32768
                   0, 0 };
      i32x4 g2 = { 0, 0, 0, 0 };
      i32x4 g3 = { 0, 0, 0, 0 };
      i32x8 g4 = { 0, 0, 0, 0, 0, 0, 0, 0 };   // extra group (6-arg toolchain)
      __builtin_amdgcn_tensor_load_to_lds(g0, g1, g2, g3, g4, 0);
#if __has_builtin(__builtin_amdgcn_s_wait_tensorcnt)
      __builtin_amdgcn_s_wait_tensorcnt(0);
#else
      asm volatile("s_wait_tensorcnt 0x0" ::: "memory");
#endif
    }
#else
    {
      const char* src = (const char*)hs;
      char* dst = (char*)shH;
#ifdef HAVE_ASYNC_LDS
#pragma unroll
      for (int kk = 0; kk < 16; ++kk) {
        const int c = ((int)threadIdx.x + (kk << 8)) << 4;
        __builtin_amdgcn_global_load_async_to_lds_b128(
            (v4i_vs*)(src + c), (v4i_vs*)(dst + c), 0, 0);
      }
#if __has_builtin(__builtin_amdgcn_s_wait_asynccnt)
      __builtin_amdgcn_s_wait_asynccnt(0);
#else
      asm volatile("s_wait_asynccnt 0" ::: "memory");
#endif
#else
#pragma unroll
      for (int kk = 0; kk < 16; ++kk) {
        const int c = (int)threadIdx.x + (kk << 8);
        ((uint4*)dst)[c] = ((const uint4*)src)[c];
      }
#endif
    }
#endif
    __syncthreads();
    // all WGs of this direction consumed old h before anyone overwrites it
    grid_barrier(ctr, target += NWG_DIR);

    // ---- 3) h @ Wh for this wave's 16x16 tile, 4 gates, ping-pong pipeline ----
    v8f acc[4] = {};
    const int amr = (mt << 4) + (lane & 15);
    const int nc  = lane & 15;
    v16bf aA = load_a_frag(shH, amr, 0, lane);             // ds_load_b128
    v16bf bA[4];
#pragma unroll
    for (int g = 0; g < 4; ++g)
      bA[g] = load_b_frag(WhT, (g << 10) + col16 + nc, 0, lane);

    for (int ko = 0; ko < DIM; ko += 64) {
      v16bf aB = load_a_frag(shH, amr, ko + 32, lane);
      v16bf bB[4];
#pragma unroll
      for (int g = 0; g < 4; ++g)
        bB[g] = load_b_frag(WhT, (g << 10) + col16 + nc, ko + 32, lane);
#pragma unroll
      for (int g = 0; g < 4; ++g) acc[g] = wmma_bf16(aA, bA[g], acc[g]);
      if (ko + 64 < DIM) {
        aA = load_a_frag(shH, amr, ko + 64, lane);
#pragma unroll
        for (int g = 0; g < 4; ++g)
          bA[g] = load_b_frag(WhT, (g << 10) + col16 + nc, ko + 64, lane);
      }
#pragma unroll
      for (int g = 0; g < 4; ++g) acc[g] = wmma_bf16(aB, bB[g], acc[g]);
    }

    // ---- 4) make the async pre-slice visible, then combine gates ----
#ifdef HAVE_ASYNC_LDS
#if __has_builtin(__builtin_amdgcn_s_wait_asynccnt)
    __builtin_amdgcn_s_wait_asynccnt(0);
#else
    asm volatile("s_wait_asynccnt 0" ::: "memory");
#endif
#endif
    __syncthreads();

    const int mrow = (mt << 4) + ((lane >> 4) << 3);
    const int col  = col16 + nc;
    const int lcol = col - (wg << 6);     // 0..63 within WG slice
#pragma unroll
    for (int r = 0; r < 8; ++r) {
      const int m = mrow + r;
      const float fg = sigmoidf_(acc[0][r] + shPre[((m << 2) + 0) * 64 + lcol]);
      const float ig = sigmoidf_(acc[1][r] + shPre[((m << 2) + 1) * 64 + lcol]);
      const float gg = tanhf    (acc[2][r] + shPre[((m << 2) + 2) * 64 + lcol]);
      const float og = sigmoidf_(acc[3][r] + shPre[((m << 2) + 3) * 64 + lcol]);
      const size_t sidx = (size_t)m * DIM + col;
      const float cn = fg * cs[sidx] + ig * gg;
      const float hn = og * tanhf(cn);
      cs[sidx] = cn;
      hs[sidx] = __float2bfloat16(hn);
      out[((size_t)m * TLEN + ts) * (2 * DIM) + ((size_t)dir << 10) + col] = hn;
    }
    // writes visible before next step's staging
    grid_barrier(ctr, target += NWG_DIR);
  }
}

// ---------------------------------------------------------------------------
// Host launcher
// ---------------------------------------------------------------------------
extern "C" void kernel_launch(void* const* d_in, const int* in_sizes, int n_in,
                              void* d_out, int out_size, void* d_ws, size_t ws_size,
                              hipStream_t stream) {
  (void)in_sizes; (void)n_in; (void)out_size; (void)ws_size;
  const float* x  = (const float*)d_in[0];
  const float* Wf = (const float*)d_in[1];
  const float* bf = (const float*)d_in[2];
  const float* Wi = (const float*)d_in[3];
  const float* bi = (const float*)d_in[4];
  const float* Wc = (const float*)d_in[5];
  const float* bc = (const float*)d_in[6];
  const float* Wo = (const float*)d_in[7];
  const float* bo = (const float*)d_in[8];
  float* out = (float*)d_out;

  char* ws = (char*)d_ws;
  // state region (zeroed every call for determinism)
  __hip_bfloat16* hstate = (__hip_bfloat16*)(ws + 0);           // 131072 B
  float*          cstate = (float*)(ws + 131072);               // 262144 B
  unsigned*       ctrs   = (unsigned*)(ws + 131072 + 262144);   // 256 B
  // bulk buffers
  __hip_bfloat16* Xb  = (__hip_bfloat16*)(ws + (1ull  << 20));  // 16 MB
  __hip_bfloat16* WxT = (__hip_bfloat16*)(ws + (17ull << 20));  //  8 MB
  __hip_bfloat16* WhT = (__hip_bfloat16*)(ws + (25ull << 20));  //  8 MB
  float*          pre = (float*)(ws + (33ull << 20));           // 128 MB

  (void)hipMemsetAsync(ws, 0, 131072 + 262144 + 256, stream);

  pack_kernel<<<32768, 256, 0, stream>>>(x, Wf, Wi, Wc, Wo, Xb, WxT, WhT);

  gemm_pre_kernel<<<4096, 256, 0, stream>>>(Xb, WxT, bf, bi, bc, bo, pre);

  lstm_recurrent_kernel<<<32, 256, 0, stream>>>(WhT, pre, hstate, cstate, ctrs, out);
}
